// LSTMAttentionDecoder_36275293782200
// MI455X (gfx1250) — matrix operand order repaired and checked by reading, not compiled, not run on hardware
//
#include <hip/hip_runtime.h>

// MI455X / gfx1250 implementation of the LSTM-attention decoder.
//
// Math shortcut (exact, from the reference): softmax over a length-1 axis is
// identically 1, so attention ctx depends only on visual_feat -> compute once.
// The hp/att branches are dead code and are skipped entirely.
//
// All GEMMs use v_wmma_f32_16x16x32_bf16 (bf16 inputs, fp32 accumulate).
// Weights are converted once to bf16 and repacked into the WMMA B-fragment
// layout (~112 MB total -> fits the 192 MB L2, so the 32 sequential timesteps
// re-read weights from L2, not HBM at 23.3 TB/s).
//
// GEMM is register-blocked over M: each wave owns a full 64-row x 16-col strip
// (4 WMMA accumulators) so each weight (B) fragment is loaded exactly once.

typedef __bf16 bf16;
typedef __attribute__((ext_vector_type(16))) __bf16 v16bf;
typedef __attribute__((ext_vector_type(8)))  float  v8f;

static inline int cdiv(long long a, int b) { return (int)((a + b - 1) / b); }

// ---- fp32 (K x N, row-major) -> bf16 packed in WMMA B-fragment tiles.
// Element order: idx = ((nt*KT + kt)*512) + lane*16 + i
//   lane 0..31, i 0..15:  n = nt*16 + (lane&15),  k = kt*32 + i + 16*(lane>>4)
__global__ void pack_w_kernel(const float* __restrict__ W, bf16* __restrict__ out,
                              int K, int N) {
  long long idx = (long long)blockIdx.x * blockDim.x + threadIdx.x;
  long long total = (long long)K * N;
  if (idx >= total) return;
  int KT = K >> 5;
  long long t = idx >> 9;
  int r = (int)(idx & 511);
  int lane = r >> 4, i = r & 15;
  int nt = (int)(t / KT), kt = (int)(t % KT);
  int k = (kt << 5) + i + ((lane >> 4) << 4);
  int n = (nt << 4) + (lane & 15);
  out[idx] = (bf16)W[(long long)k * N + n];
}

// ---- WMMA GEMM, M fixed at 64 (4 M-tiles per wave):
//   C[64,N] = A1[64,K1]@W1 (+ A2[64,K2]@W2) + b1 (+ b2)
// A row-major bf16; W packed by pack_w_kernel; C fp32 with row stride ldc.
// One wave per 16-column strip: waves = N/16. K multiples of 32.
__global__ void gemm_wmma_m64_kernel(const bf16* __restrict__ A1, const bf16* __restrict__ W1,
                                     int K1, int lda1,
                                     const bf16* __restrict__ A2, const bf16* __restrict__ W2,
                                     int K2, int lda2,
                                     const float* __restrict__ b1, const float* __restrict__ b2,
                                     float* __restrict__ C, long long ldc, int N) {
  int wid = (int)((blockIdx.x * blockDim.x + threadIdx.x) >> 5);
  int ntiles = N >> 4;
  if (wid >= ntiles) return;                    // uniform per wave: EXEC all-ones
  int nt = wid;
  int lane = threadIdx.x & 31;
  int half = lane >> 4, lm = lane & 15;

  v8f acc[4];
#pragma unroll
  for (int mt = 0; mt < 4; ++mt) acc[mt] = (v8f){0.f, 0.f, 0.f, 0.f, 0.f, 0.f, 0.f, 0.f};

  {
    int KT = K1 >> 5;
    // A frag: lane holds row M = mt*16+lm; element i -> k = kb + i + 16*half
    const bf16* arow = A1 + (long long)lm * lda1 + (half << 4);
    const bf16* wp   = W1 + (((long long)nt * KT) << 9) + (lane << 4);
    for (int kt = 0; kt < KT; ++kt) {
      int kb = kt << 5;
      v16bf bfrag = *(const v16bf*)wp;          // 2x coalesced b128 per lane
      if (kt + 1 < KT) __builtin_prefetch(wp + 512, 0, 3);   // global_prefetch_b8
      wp += 512;
      v16bf a[4];
#pragma unroll
      for (int mt = 0; mt < 4; ++mt) {
        const bf16* ar = arow + (long long)(mt << 4) * lda1;
#pragma unroll
        for (int i = 0; i < 16; ++i) a[mt][i] = ar[kb + i];
      }
#pragma unroll
      for (int mt = 0; mt < 4; ++mt)
        acc[mt] = __builtin_amdgcn_wmma_f32_16x16x32_bf16(false, a[mt], false, bfrag,
                                                          (short)0, acc[mt], false, false);
    }
  }
  if (A2) {
    int KT = K2 >> 5;
    const bf16* arow = A2 + (long long)lm * lda2 + (half << 4);
    const bf16* wp   = W2 + (((long long)nt * KT) << 9) + (lane << 4);
    for (int kt = 0; kt < KT; ++kt) {
      int kb = kt << 5;
      v16bf bfrag = *(const v16bf*)wp;
      if (kt + 1 < KT) __builtin_prefetch(wp + 512, 0, 3);
      wp += 512;
      v16bf a[4];
#pragma unroll
      for (int mt = 0; mt < 4; ++mt) {
        const bf16* ar = arow + (long long)(mt << 4) * lda2;
#pragma unroll
        for (int i = 0; i < 16; ++i) a[mt][i] = ar[kb + i];
      }
#pragma unroll
      for (int mt = 0; mt < 4; ++mt)
        acc[mt] = __builtin_amdgcn_wmma_f32_16x16x32_bf16(false, a[mt], false, bfrag,
                                                          (short)0, acc[mt], false, false);
    }
  }

  int n = (nt << 4) + lm;
  float bias = 0.f;
  if (b1) bias += b1[n];
  if (b2) bias += b2[n];
#pragma unroll
  for (int mt = 0; mt < 4; ++mt) {
    int mbase = (mt << 4) + (half << 3);        // C/D layout: VGPR r -> M = 8*half + r
#pragma unroll
    for (int r = 0; r < 8; ++r)
      C[(long long)(mbase + r) * ldc + n] = acc[mt][r] + bias;
  }
}

// ---- LayerNorm + activation. One block per row. act: 0=none, 1=relu, 2=tanh.
// Optionally writes bf16 (strided) and/or fp32 outputs.
__global__ void ln_act_kernel(const float* __restrict__ X, int N,
                              const float* __restrict__ g, const float* __restrict__ b,
                              int act, bf16* outB, long long ldoB,
                              float* outF, long long ldoF) {
  int row = blockIdx.x;
  const float* x = X + (long long)row * N;
  float s = 0.f, ss = 0.f;
  for (int i = threadIdx.x; i < N; i += blockDim.x) { float v = x[i]; s += v; ss += v * v; }
#pragma unroll
  for (int off = 16; off > 0; off >>= 1) {
    s  += __shfl_down(s,  off, 32);
    ss += __shfl_down(ss, off, 32);
  }
  __shared__ float sb[18];
  int wv = threadIdx.x >> 5;
  if ((threadIdx.x & 31) == 0) { sb[wv] = s; sb[8 + wv] = ss; }
  __syncthreads();
  if (threadIdx.x == 0) {
    float ts = 0.f, tss = 0.f;
    int nw = (int)((blockDim.x + 31) >> 5);
    for (int i = 0; i < nw; ++i) { ts += sb[i]; tss += sb[8 + i]; }
    float mean = ts / N;
    float var  = tss / N - mean * mean;
    sb[16] = mean;
    sb[17] = rsqrtf(var + 1e-5f);
  }
  __syncthreads();
  float mean = sb[16], rstd = sb[17];
  for (int i = threadIdx.x; i < N; i += blockDim.x) {
    float y = (x[i] - mean) * rstd * g[i] + b[i];
    if (act == 1) y = fmaxf(y, 0.f);
    else if (act == 2) y = tanhf(y);
    if (outB) outB[(long long)row * ldoB + i] = (bf16)y;
    if (outF) outF[(long long)row * ldoF + i] = y;
  }
}

// ---- LSTM cell elementwise: gates (B,4H) order i,f,g,o
__global__ void lstm_cell_kernel(const float* __restrict__ gates,
                                 float* __restrict__ c, float* __restrict__ h,
                                 bf16* __restrict__ hb, bf16* __restrict__ xh,
                                 int n, int Hc) {
  int i = blockIdx.x * blockDim.x + threadIdx.x;
  if (i >= n) return;
  int bb = i / Hc, hh = i % Hc;
  const float* gr = gates + (long long)bb * (4 * Hc);
  float gi = gr[hh], gf = gr[Hc + hh], gg = gr[2 * Hc + hh], go = gr[3 * Hc + hh];
  float si = 1.f / (1.f + __expf(-gi));
  float sf = 1.f / (1.f + __expf(-gf));
  float so = 1.f / (1.f + __expf(-go));
  float cn = sf * c[i] + si * tanhf(gg);
  float hn = so * tanhf(cn);
  c[i] = cn;
  h[i] = hn;
  hb[i] = (bf16)hn;
  xh[i] = (bf16)hn;
}

// ---- embedding gather -> (T,B,E) bf16
__global__ void gather_emb_kernel(const int* __restrict__ ids, const float* __restrict__ emb,
                                  bf16* __restrict__ out, int Bc, int Tc, int Ec) {
  long long idx = (long long)blockIdx.x * blockDim.x + threadIdx.x;
  if (idx >= (long long)Tc * Bc * Ec) return;
  int e = (int)(idx % Ec);
  long long r = idx / Ec;
  int bb = (int)(r % Bc);
  int t  = (int)(r / Bc);
  int id = ids[bb * Tc + t];
  out[idx] = (bf16)emb[(long long)id * Ec + e];
}

__global__ void cvt_bf16_kernel(const float* __restrict__ x, bf16* __restrict__ y, int n) {
  int i = blockIdx.x * blockDim.x + threadIdx.x;
  if (i < n) y[i] = (bf16)x[i];
}

// copy per-step embedding slice into cols [0,E) of the concat-x buffer
__global__ void set_x_emb_kernel(const bf16* __restrict__ embT, bf16* __restrict__ x,
                                 int Bc, int Ec, int ldx) {
  int i = blockIdx.x * blockDim.x + threadIdx.x;
  if (i >= Bc * Ec) return;
  int bb = i / Ec, e = i % Ec;
  x[(long long)bb * ldx + e] = embT[i];
}

// tile (B,H) state into L layer slots (fp32 + optional bf16)
__global__ void tile_state_kernel(const float* __restrict__ src, float* __restrict__ dstF,
                                  bf16* __restrict__ dstB, int n, int Lc) {
  int i = blockIdx.x * blockDim.x + threadIdx.x;
  if (i >= n) return;
  float v = src[i];
  for (int l = 0; l < Lc; ++l) {
    dstF[(long long)l * n + i] = v;
    if (dstB) dstB[(long long)l * n + i] = (bf16)v;
  }
}

__global__ void copyf_kernel(const float* __restrict__ s, float* __restrict__ d, int n) {
  int i = blockIdx.x * blockDim.x + threadIdx.x;
  if (i < n) d[i] = s[i];
}

extern "C" void kernel_launch(void* const* d_in, const int* in_sizes, int n_in,
                              void* d_out, int out_size, void* d_ws, size_t ws_size,
                              hipStream_t stream) {
  const int Bb = 64, T = 32, V = 10000, E = 768, H = 1024, L = 5, F = 512;
  const int EH = E + H;                          // 1792

  const int*   word_ids  = (const int*)d_in[0];
  const float* feat      = (const float*)d_in[1];
  const float* embedding = (const float*)d_in[2];
  float* out = (float*)d_out;
  auto f = [&](int i) { return (const float*)d_in[i]; };

  // bump allocator over workspace (needs ~122 MB: 112 MB bf16 weights + acts)
  char* wsb = (char*)d_ws;
  size_t off = 0;
  auto alloc = [&](size_t bytes) -> char* {
    char* p = wsb + off;
    off = (off + bytes + 255) & ~(size_t)255;
    return p;
  };

  // ---- weight pack specs: {d_in index, K, N}; hp/att branches skipped (dead)
  struct WSpec { int idx, K, N; };
  const WSpec wspec[21] = {
    {11,  512, 1536},  // 0  attn.fp.l1
    {15, 1536,  768},  // 1  attn.fp.l2
    {27,  768, 1024},  // 2  attn.op.l1
    {31, 1024, 1024},  // 3  attn.op.l2
    {35,  512, 1024},  // 4  h_init.l1
    {39, 1024, 1024},  // 5  h_init.l2
    {43,  512, 1024},  // 6  c_init.l1
    {47, 1024, 1024},  // 7  c_init.l2
    {51, 1792, 4096},  // 8  lstm0.wih
    {52, 1024, 4096},  // 9  lstm0.whh
    {55, 1024, 4096},  // 10 lstm1.wih
    {56, 1024, 4096},  // 11 lstm1.whh
    {59, 1024, 4096},  // 12 lstm2.wih
    {60, 1024, 4096},  // 13 lstm2.whh
    {63, 1024, 4096},  // 14 lstm3.wih
    {64, 1024, 4096},  // 15 lstm3.whh
    {67, 1024, 4096},  // 16 lstm4.wih
    {68, 1024, 4096},  // 17 lstm4.whh
    {71, 1024, 1024},  // 18 out.l1
    {75, 1024,  512},  // 19 out.l2
    {79,  512,10000},  // 20 out.l3
  };
  bf16* Wp[21];
  for (int i = 0; i < 21; ++i) {
    long long elems = (long long)wspec[i].K * wspec[i].N;
    Wp[i] = (bf16*)alloc((size_t)elems * 2);
    pack_w_kernel<<<cdiv(elems, 256), 256, 0, stream>>>(f(wspec[i].idx), Wp[i],
                                                        wspec[i].K, wspec[i].N);
  }

  // ---- activation buffers
  bf16*  embbf  = (bf16*)alloc((size_t)T * Bb * E * 2);
  bf16*  featbf = (bf16*)alloc((size_t)Bb * F * 2);
  bf16*  xbuf   = (bf16*)alloc((size_t)Bb * EH * 2);    // [emb_t | ctx]
  bf16*  xh     = (bf16*)alloc((size_t)Bb * H * 2);     // inter-layer h (bf16)
  float* hf     = (float*)alloc((size_t)L * Bb * H * 4);
  float* cf     = (float*)alloc((size_t)L * Bb * H * 4);
  bf16*  hbf    = (bf16*)alloc((size_t)L * Bb * H * 2);
  float* gates  = (float*)alloc((size_t)Bb * 4 * H * 4);
  float* t32    = (float*)alloc((size_t)Bb * 1536 * 4);
  bf16*  tb16a  = (bf16*)alloc((size_t)Bb * 1536 * 2);
  bf16*  tb16b  = (bf16*)alloc((size_t)Bb * 1536 * 2);
  float* s0     = (float*)alloc((size_t)Bb * H * 4);
  (void)ws_size; (void)in_sizes; (void)n_in; (void)out_size;

  auto gemm1 = [&](const bf16* A, const bf16* W, int K, int lda, const float* bias,
                   float* C, long long ldc, int N) {
    int waves = N / 16;
    gemm_wmma_m64_kernel<<<cdiv((long long)waves * 32, 256), 256, 0, stream>>>(
        A, W, K, lda, nullptr, nullptr, 0, 0, bias, nullptr, C, ldc, N);
  };
  auto gemm2 = [&](const bf16* A1, const bf16* W1, int K1, int lda1,
                   const bf16* A2, const bf16* W2, int K2, int lda2,
                   const float* b1, const float* b2, float* C, long long ldc, int N) {
    int waves = N / 16;
    gemm_wmma_m64_kernel<<<cdiv((long long)waves * 32, 256), 256, 0, stream>>>(
        A1, W1, K1, lda1, A2, W2, K2, lda2, b1, b2, C, ldc, N);
  };
  auto ln = [&](const float* X, int N, int gi, int bi, int act,
                bf16* oB, long long ldoB, float* oF, long long ldoF) {
    ln_act_kernel<<<Bb, 256, 0, stream>>>(X, N, f(gi), f(bi), act, oB, ldoB, oF, ldoF);
  };

  // ---- one-time setup
  gather_emb_kernel<<<cdiv((long long)T * Bb * E, 256), 256, 0, stream>>>(
      word_ids, embedding, embbf, Bb, T, E);
  cvt_bf16_kernel<<<cdiv(Bb * F, 256), 256, 0, stream>>>(feat, featbf, Bb * F);

  // ctx = op(fp(feat))   (softmax over singleton axis == 1 -> h-independent)
  gemm1(featbf, Wp[0], 512, F, f(12), t32, 1536, 1536);
  ln(t32, 1536, 13, 14, 1, tb16a, 1536, nullptr, 0);
  gemm1(tb16a, Wp[1], 1536, 1536, f(16), t32, 768, 768);
  ln(t32, 768, 17, 18, 1, tb16b, 768, nullptr, 0);
  gemm1(tb16b, Wp[2], 768, 768, f(28), t32, 1024, 1024);
  ln(t32, 1024, 29, 30, 1, tb16a, 1024, nullptr, 0);
  gemm1(tb16a, Wp[3], 1024, 1024, f(32), t32, 1024, 1024);
  ln(t32, 1024, 33, 34, 0, xbuf + E, (long long)EH, nullptr, 0);  // ctx -> x[:,768:]

  // h0 = tanh(LN(relu(LN(feat@l1))@l2)), tiled to all L layers
  gemm1(featbf, Wp[4], 512, F, f(36), t32, 1024, 1024);
  ln(t32, 1024, 37, 38, 1, tb16a, 1024, nullptr, 0);
  gemm1(tb16a, Wp[5], 1024, 1024, f(40), t32, 1024, 1024);
  ln(t32, 1024, 41, 42, 2, nullptr, 0, s0, 1024);
  tile_state_kernel<<<cdiv(Bb * H, 256), 256, 0, stream>>>(s0, hf, hbf, Bb * H, L);
  // c0
  gemm1(featbf, Wp[6], 512, F, f(44), t32, 1024, 1024);
  ln(t32, 1024, 45, 46, 1, tb16a, 1024, nullptr, 0);
  gemm1(tb16a, Wp[7], 1024, 1024, f(48), t32, 1024, 1024);
  ln(t32, 1024, 49, 50, 2, nullptr, 0, s0, 1024);
  tile_state_kernel<<<cdiv(Bb * H, 256), 256, 0, stream>>>(s0, cf, nullptr, Bb * H, L);

  // ---- time loop (sequential recurrence; weights stay hot in L2)
  for (int t = 0; t < T; ++t) {
    set_x_emb_kernel<<<cdiv(Bb * E, 256), 256, 0, stream>>>(
        embbf + (long long)t * Bb * E, xbuf, Bb, E, EH);

    // layer 0: gates = x@wih0 + h0@whh0 + bih0 + bhh0
    gemm2(xbuf, Wp[8], 1792, 1792, hbf, Wp[9], 1024, 1024, f(53), f(54),
          gates, 4096, 4096);
    lstm_cell_kernel<<<cdiv(Bb * H, 256), 256, 0, stream>>>(
        gates, cf, hf, hbf, xh, Bb * H, H);

    for (int l = 1; l < L; ++l) {
      int wi = 8 + 2 * l, bi = 51 + 4 * l;
      gemm2(xh, Wp[wi], 1024, 1024, hbf + (long long)l * Bb * H, Wp[wi + 1],
            1024, 1024, f(bi + 2), f(bi + 3), gates, 4096, 4096);
      lstm_cell_kernel<<<cdiv(Bb * H, 256), 256, 0, stream>>>(
          gates, cf + (long long)l * Bb * H, hf + (long long)l * Bb * H,
          hbf + (long long)l * Bb * H, xh, Bb * H, H);
    }

    // output head -> logits[:, t, :]
    gemm1(xh, Wp[18], 1024, 1024, f(72), t32, 1024, 1024);
    ln(t32, 1024, 73, 74, 1, tb16a, 1024, nullptr, 0);
    gemm1(tb16a, Wp[19], 1024, 1024, f(76), t32, 512, 512);
    ln(t32, 512, 77, 78, 1, tb16b, 512, nullptr, 0);
    gemm1(tb16b, Wp[20], 512, 512, f(80), out + (long long)t * V, (long long)T * V, V);
  }

  // ---- final h, c states
  copyf_kernel<<<cdiv(L * Bb * H, 256), 256, 0, stream>>>(
      hf, out + (long long)Bb * T * V, L * Bb * H);
  copyf_kernel<<<cdiv(L * Bb * H, 256), 256, 0, stream>>>(
      cf, out + (long long)Bb * T * V + (long long)L * Bb * H, L * Bb * H);
}
